// RecurTreeGen_19533511262867
// MI455X (gfx1250) — matrix operand order repaired and checked by reading, not compiled, not run on hardware
//
#include <hip/hip_runtime.h>
#include <stdint.h>

// ---------------- problem constants (from reference) ----------------
#define D        128
#define NLEAVES  (64 * 4096)
#define NLEVELS  12

// GEMM tiling: C = h_cat[16 x 256] * W[256 x 640]
#define KT   8          // k-steps of 32 (K = 256)
#define NT   40         // n-tiles of 16 (N = 640)
#define MT   16         // rows per block
#define ROWPAD 264      // 256 + 8 ushort pad -> 528B row stride (bank rotate)

typedef __attribute__((ext_vector_type(16))) __bf16 v16bf;
typedef __attribute__((ext_vector_type(8)))  float  v8f;

__device__ __forceinline__ unsigned short f2bf(float f) {
    union { float f; unsigned int u; } c; c.f = f;
    unsigned int u = c.u;
    u += 0x7FFFu + ((u >> 16) & 1u);          // round-to-nearest-even
    return (unsigned short)(u >> 16);
}

__device__ __forceinline__ float sigm(float x) {
    return 1.0f / (1.0f + __expf(-x));
}

// ---------------------------------------------------------------
// Pack W = [U_iou | Uf_W] (f32, 256 x 640) into bf16 B-fragment order:
// wp[((kk*NT + nt)*32 + lane)*16 + j]
//   lane 0-15 : N = lane,     K = kk*32 + {0..7, 16..23}  (j = 0..15)
//   lane 16-31: N = lane-16,  K = kk*32 + {8..15, 24..31}
// (matches V_WMMA_F32_16X16X32_BF16 B-matrix 32x16 per-lane layout)
// ---------------------------------------------------------------
__global__ void pack_weights(const float* __restrict__ U_iou,
                             const float* __restrict__ Uf_W,
                             unsigned short* __restrict__ wp) {
    int idx  = blockIdx.x * blockDim.x + threadIdx.x;   // < KT*NT*32*16
    int j    = idx & 15;
    int lane = (idx >> 4) & 31;
    int t    = idx >> 9;                                 // kk*NT + nt
    int nt   = t % NT;
    int kk   = t / NT;
    int K    = kk * 32 + (lane >> 4) * 8 + (j & 7) + (j >> 3) * 16;
    int col  = nt * 16 + (lane & 15);
    float v  = (col < 3 * D) ? U_iou[(size_t)K * (3 * D) + col]
                             : Uf_W [(size_t)K * (2 * D) + (col - 3 * D)];
    wp[idx] = f2bf(v);
}

// ---------------------------------------------------------------
// One Fenwick merge level: 16 parent nodes per 256-thread block.
// ---------------------------------------------------------------
__global__ __launch_bounds__(256) void tree_lstm_level(
    const float* __restrict__ hin, const float* __restrict__ cin,
    const unsigned short* __restrict__ wp,
    const float* __restrict__ b_iou, const float* __restrict__ uf_b,
    float* __restrict__ hout, float* __restrict__ cout, int last)
{
    __shared__ unsigned short sA[MT * ROWPAD];   // h_cat tile, bf16, padded rows

    const int blk  = blockIdx.x;
    const int tid  = threadIdx.x;

    // ---- stage A: h_cat[row][0:256] = [h_left | h_right], f32 -> bf16 ----
    {
        int row = tid >> 4;                      // 0..15
        int seg = tid & 15;                      // 16 cols each
        size_t g2 = ((size_t)blk * 16 + row) * 2;          // left child
        const float* src = (seg < 8) ? (hin + g2 * D + seg * 16)
                                     : (hin + (g2 + 1) * D + (seg - 8) * 16);
        unsigned short* dst = &sA[row * ROWPAD + seg * 16];
        #pragma unroll
        for (int q = 0; q < 2; ++q) {
            float4 v0 = ((const float4*)src)[q * 2 + 0];
            float4 v1 = ((const float4*)src)[q * 2 + 1];
            uint4 w;
            w.x = (unsigned)f2bf(v0.x) | ((unsigned)f2bf(v0.y) << 16);
            w.y = (unsigned)f2bf(v0.z) | ((unsigned)f2bf(v0.w) << 16);
            w.z = (unsigned)f2bf(v1.x) | ((unsigned)f2bf(v1.y) << 16);
            w.w = (unsigned)f2bf(v1.z) | ((unsigned)f2bf(v1.w) << 16);
            ((uint4*)dst)[q] = w;
        }
    }
    __syncthreads();

    const int wave  = tid >> 5;
    const int lane  = tid & 31;
    const int rowA  = lane & 15;
    const int halfA = lane >> 4;

    v8f acc[5] = {};   // i, o, u, f1, f2 column tiles (16x16 f32 each)

    // ---- GEMM: 8 k-steps x 5 n-tiles of V_WMMA_F32_16X16X32_BF16 ----
    #pragma unroll
    for (int kk = 0; kk < KT; ++kk) {
        // A fragment (16x32 bf16, ISA per-lane layout):
        //   lane<16 : row=lane,    elems 0-7 -> K=kk*32+0..7,  8-15 -> +16..23
        //   lane>=16: row=lane-16, elems 0-7 -> K=kk*32+8..15, 8-15 -> +24..31
        union { uint4 q[2]; v16bf v; } A;
        const unsigned short* ap = &sA[rowA * ROWPAD + kk * 32 + halfA * 8];
        A.q[0] = *(const uint4*)(ap);
        A.q[1] = *(const uint4*)(ap + 16);
        #pragma unroll
        for (int j = 0; j < 5; ++j) {
            int nt = wave + j * 8;
            union { uint4 q[2]; v16bf v; } B;
            const uint4* bp =
                (const uint4*)(wp + (((size_t)kk * NT + nt) * 32 + lane) * 16);
            B.q[0] = bp[0];
            B.q[1] = bp[1];
            acc[j] = __builtin_amdgcn_wmma_f32_16x16x32_bf16(
                         false, A.v, false, B.v, (short)0, acc[j], false, false);
        }
    }

    // ---- elementwise tree-LSTM cell, fully register-local per wave ----
    // wave w owns output columns col = 16w + (lane&15) for gates i,o,u,f1,f2
    const int col = wave * 16 + (lane & 15);
    const float bi  = b_iou[col];
    const float bo  = b_iou[D + col];
    const float bu  = b_iou[2 * D + col];
    const float bf1 = uf_b[col];
    const float bf2 = uf_b[D + col];
    const int rbase = (lane >> 4) * 8;   // C/D layout: M = v + 8*(lane>=16)

    #pragma unroll
    for (int v = 0; v < 8; ++v) {
        int r = v + rbase;
        size_t g = (size_t)blk * 16 + r;
        float iv = sigm(acc[0][v] + bi);
        float ov = sigm(acc[1][v] + bo);
        float uv = tanhf(acc[2][v] + bu);
        float f1 = sigm(acc[3][v] + bf1);
        float f2 = sigm(acc[4][v] + bf2);
        float cl = cin[(2 * g) * D + col];
        float cr = cin[(2 * g + 1) * D + col];
        float cn = iv * uv + f1 * cl + f2 * cr;
        float hn = ov * tanhf(cn);
        if (last) {   // root: d_out layout [BATCH, 2D] = [h | c]
            hout[g * (2 * D) + col]     = hn;
            hout[g * (2 * D) + D + col] = cn;
        } else {
            hout[g * D + col] = hn;
            cout[g * D + col] = cn;
        }
    }
}

// ---------------------------------------------------------------
extern "C" void kernel_launch(void* const* d_in, const int* in_sizes, int n_in,
                              void* d_out, int out_size, void* d_ws, size_t ws_size,
                              hipStream_t stream) {
    (void)in_sizes; (void)n_in; (void)out_size; (void)ws_size;

    const float* h_bot = (const float*)d_in[0];
    const float* c_bot = (const float*)d_in[1];
    const float* U_iou = (const float*)d_in[2];
    const float* b_iou = (const float*)d_in[3];
    const float* Uf_W  = (const float*)d_in[4];
    const float* Uf_b  = (const float*)d_in[5];
    float* out = (float*)d_out;

    // workspace layout
    const size_t WP_ELEMS = (size_t)KT * NT * 32 * 16;       // 163840 bf16
    unsigned short* wp = (unsigned short*)d_ws;
    float* fbase = (float*)((char*)d_ws + WP_ELEMS * 2);     // 327,680 B in
    const size_t PING_N = NLEAVES / 2;                       // 131072 nodes
    const size_t PONG_N = NLEAVES / 4;                       // 65536 nodes
    float* pingH = fbase;
    float* pingC = pingH + PING_N * D;
    float* pongH = pingC + PING_N * D;
    float* pongC = pongH + PONG_N * D;

    // 1) pack shared weights into bf16 WMMA B-fragment order
    pack_weights<<<(int)(WP_ELEMS / 256), 256, 0, stream>>>(U_iou, Uf_W, wp);

    // 2) 12 dependent merge levels
    const float* hin = h_bot;
    const float* cin = c_bot;
    for (int L = 0; L < NLEVELS; ++L) {
        int n_out  = NLEAVES >> (L + 1);
        int blocks = n_out / MT;
        int lastL  = (L == NLEVELS - 1);
        float *ho, *co;
        if (lastL)            { ho = out;   co = nullptr; }
        else if ((L & 1) == 0){ ho = pingH; co = pingC;   }
        else                  { ho = pongH; co = pongC;   }
        tree_lstm_level<<<blocks, 256, 0, stream>>>(hin, cin, wp, b_iou, Uf_b,
                                                    ho, co, lastL);
        hin = ho; cin = co;
    }
}
